// VTIRTSingleKC_8469675507927
// MI455X (gfx1250) — matrix operand order repaired and checked by reading, not compiled, not run on hardware
//
#include <hip/hip_runtime.h>
#include <hip/hip_bf16.h>
#include <math.h>

typedef __attribute__((ext_vector_type(16))) _Float16 v16h;
typedef __attribute__((ext_vector_type(8)))  float    v8f;

#define U_DIM 2048
#define T_DIM 512
#define N_ROWS (U_DIM * T_DIM)

__device__ __forceinline__ float gelu_exact(float x) {
    // exact GELU: 0.5*x*(1+erf(x/sqrt(2)))
    return 0.5f * x * (1.0f + erff(x * 0.70710678118654752440f));
}

// ---------------- kernel 1: reparameterize diff/disc (Q elements) ----------
__global__ void reparam_kernel(const float* __restrict__ mu,
                               const float* __restrict__ logvar,
                               const float* __restrict__ eps,
                               float* __restrict__ out, int Q) {
    int i = blockIdx.x * blockDim.x + threadIdx.x;
    if (i < Q) out[i] = mu[i] + expf(0.5f * logvar[i]) * eps[i];
}

// ---------------- kernel 2: transpose mask/eps_ab to [T][U] ----------------
__global__ void transpose_kernel(const int* __restrict__ mask,
                                 const float* __restrict__ eps,
                                 float* __restrict__ maskT,
                                 float* __restrict__ epsT) {
    int idx = blockIdx.x * blockDim.x + threadIdx.x;  // over U*T, row-major (u,t)
    int u = idx >> 9;             // / T_DIM
    int t = idx & (T_DIM - 1);
    int o = t * U_DIM + u;
    maskT[o] = (float)mask[idx];
    epsT[o]  = eps[idx];
}

// ---------------- kernel 3: fused MLP with WMMA ----------------------------
// Each wave computes a 16-row tile of the MLP:
//   h1 = gelu(inp @ W1 + b1)            (K=3, VALU, built directly into A frags)
//   h2 = gelu(h1 @ W2 + b2)             (128x128, v_wmma_f32_16x16x32_f16 x32)
//   poten = h2 @ W3 + b3                (N=2, VALU on C-frag layout + shfl_xor)
__global__ void __launch_bounds__(256) mlp_wmma_kernel(
    const int*   __restrict__ q_id, const float* __restrict__ resp,
    const float* __restrict__ diff, const float* __restrict__ disc,
    const float* __restrict__ W1,   const float* __restrict__ b1,
    const float* __restrict__ W2,   const float* __restrict__ b2,
    const float* __restrict__ W3,   const float* __restrict__ b3,
    float* __restrict__ muT, float* __restrict__ lmdaT)
{
    // W2 pre-converted to f16, fragment-major: [kchunk][ntile][lane][16 halves]
    __shared__ __align__(32) _Float16 sB[4][8][32][16];   // 32 KB
    __shared__ float sW1[3 * 128];
    __shared__ float sb1[128];
    __shared__ float sb2[128];
    __shared__ float sW3[256];
    __shared__ float sb3[2];

    int tid = threadIdx.x;
    // Fill B fragments. B (32x16 f16 per fragment) layout: lanes 0-15 hold
    // K={0..7,16..23} of column N=lane, lanes 16-31 hold K={8..15,24..31} of column N=lane-16.
    for (int idx = tid; idx < 4 * 8 * 32 * 16; idx += 256) {
        int e = idx & 15;
        int l = (idx >> 4) & 31;
        int n = (idx >> 9) & 7;
        int c = idx >> 12;
        int col = n * 16 + (l & 15);
        int k   = 32 * c + ((l & 16) ? 8 : 0) + e + ((e >= 8) ? 8 : 0);
        sB[c][n][l][e] = (_Float16)W2[k * 128 + col];
    }
    for (int i = tid; i < 384; i += 256) sW1[i] = W1[i];
    if (tid < 128) { sb1[tid] = b1[tid]; sb2[tid] = b2[tid]; }
    for (int i = tid; i < 256; i += 256) sW3[i] = W3[i];
    if (tid < 2) sb3[tid] = b3[tid];
    __syncthreads();

    int lane = tid & 31;
    int wv   = tid >> 5;
    int m    = lane & 15;                // A-matrix row this lane contributes
    int off  = (lane < 16) ? 0 : 8;      // A layout K-offset for lane half
    int jl   = lane & 15;                // C layout: N = 16*n + jl

    const int nTiles = N_ROWS / 16;
    for (int tile = blockIdx.x * 8 + wv; tile < nTiles; tile += gridDim.x * 8) {
        int r0  = tile << 4;
        int row = r0 + m;
        int q   = q_id[row];
        float x0 = diff[q];
        float x1 = disc[q];
        float x2 = resp[row];

        // Layer 1 + GELU, packed into A fragments.
        // A f16 16x32 layout: low lanes hold K in {0..7, 16..23} of the chunk,
        // high lanes K in {8..15, 24..31}; element e<8 -> K=32c+off+e,
        // e>=8 -> K=32c+16+off+(e-8).
        v16h a[4];
        #pragma unroll
        for (int c = 0; c < 4; ++c) {
            #pragma unroll
            for (int e = 0; e < 16; ++e) {
                int j = 32 * c + off + e + ((e >= 8) ? 8 : 0);
                float h = fmaf(x0, sW1[j],
                          fmaf(x1, sW1[128 + j],
                          fmaf(x2, sW1[256 + j], sb1[j])));
                a[c][e] = (_Float16)gelu_exact(h);
            }
        }

        float s0[8], s1[8];
        #pragma unroll
        for (int p = 0; p < 8; ++p) { s0[p] = 0.f; s1[p] = 0.f; }

        for (int n = 0; n < 8; ++n) {
            v8f acc = {};
            #pragma unroll
            for (int c = 0; c < 4; ++c) {
                v16h bf = *(const v16h*)&sB[c][n][lane][0];
                acc = __builtin_amdgcn_wmma_f32_16x16x32_f16(
                        false, a[c], false, bf, (short)0, acc, false, false);
            }
            // C layout: VGPR p, low lanes -> M=p, N=16n+lane;
            //           high lanes -> M=8+p, N=16n+(lane-16).
            int j = n * 16 + jl;
            float w30 = sW3[2 * j];
            float w31 = sW3[2 * j + 1];
            float bb  = sb2[j];
            #pragma unroll
            for (int p = 0; p < 8; ++p) {
                float g = gelu_exact(acc[p] + bb);
                s0[p] = fmaf(g, w30, s0[p]);
                s1[p] = fmaf(g, w31, s1[p]);
            }
        }

        // Reduce the 16-lane N partition (stays inside each lane half).
        #pragma unroll
        for (int xm = 1; xm < 16; xm <<= 1) {
            #pragma unroll
            for (int p = 0; p < 8; ++p) {
                s0[p] += __shfl_xor(s0[p], xm, 32);
                s1[p] += __shfl_xor(s1[p], xm, 32);
            }
        }

        if ((lane & 15) < 8) {
            int p    = lane & 7;
            int mrow = p + ((lane < 16) ? 0 : 8);
            int row2 = r0 + mrow;
            int u = row2 >> 9;
            int t = row2 & (T_DIM - 1);
            int o = t * U_DIM + u;
            float mu = s0[p] + sb3[0];
            float lm = fminf(expf(-(s1[p] + sb3[1])), 1e32f);
            muT[o]   = mu;
            lmdaT[o] = lm;
        }
    }
}

// ---------------- kernel 4: backward + forward scans (per user) ------------
__global__ void scan_kernel(const float* __restrict__ muT,
                            const float* __restrict__ lmdaT,
                            const float* __restrict__ maskT,
                            const float* __restrict__ epsT,
                            float* __restrict__ A_ws, float* __restrict__ B_ws,
                            float* __restrict__ abT) {
    int u = blockIdx.x * blockDim.x + threadIdx.x;
    if (u >= U_DIM) return;
    // lth1 = lth2 = 1 since STD_THETA = 1
    float a = 0.f, b = 0.f;
    for (int t = T_DIM - 1; t >= 0; --t) {
        int o = t * U_DIM + u;
        float l = lmdaT[o], m = muT[o];
        bool mk = maskT[o] != 0.f;
        A_ws[o] = a;               // scan outputs the pre-update carry
        B_ws[o] = b;
        float denom = l + a;
        float b_new = (l * m + a * b) / denom;
        float a_new = denom / (1.f + denom);
        if (mk) { a = a_new; b = b_new; }
    }
    float ab = 0.f;
    for (int t = 0; t < T_DIM; ++t) {
        int o = t * U_DIM + u;
        float l = lmdaT[o], m = muT[o];
        bool mk = maskT[o] != 0.f;
        float an = A_ws[o], bn = B_ws[o], e = epsT[o];
        float denom = 1.f + l + an;
        float mu_t  = (ab + l * m + an * bn) / denom;
        float std_t = sqrtf(1.f - an);
        if (mk) ab = mu_t + std_t * e;
        abT[o] = ab;
    }
}

// ---------------- kernel 5: logits + output layout -------------------------
__global__ void finalize_kernel(const float* __restrict__ abT,
                                const int* __restrict__ q_id,
                                const float* __restrict__ diff,
                                const float* __restrict__ disc,
                                float* __restrict__ out) {
    int idx = blockIdx.x * blockDim.x + threadIdx.x;  // (u,t) row-major
    int u = idx >> 9;
    int t = idx & (T_DIM - 1);
    float ab = abT[t * U_DIM + u];
    int q = q_id[idx];
    out[idx]          = disc[q] * (ab - diff[q]);  // trial_logits
    out[N_ROWS + idx] = ab;                        // trial_ability
}

extern "C" void kernel_launch(void* const* d_in, const int* in_sizes, int n_in,
                              void* d_out, int out_size, void* d_ws, size_t ws_size,
                              hipStream_t stream) {
    const int*   mask     = (const int*)  d_in[0];
    const int*   q_id     = (const int*)  d_in[1];
    const float* resp     = (const float*)d_in[2];
    const float* diff_mu  = (const float*)d_in[3];
    const float* diff_lv  = (const float*)d_in[4];
    const float* disc_mu  = (const float*)d_in[5];
    const float* disc_lv  = (const float*)d_in[6];
    const float* W1       = (const float*)d_in[7];
    const float* b1       = (const float*)d_in[8];
    const float* W2       = (const float*)d_in[9];
    const float* b2       = (const float*)d_in[10];
    const float* W3       = (const float*)d_in[11];
    const float* b3       = (const float*)d_in[12];
    const float* eps_diff = (const float*)d_in[13];
    const float* eps_disc = (const float*)d_in[14];
    const float* eps_ab   = (const float*)d_in[15];
    int Q = in_sizes[3];
    int QP = ((Q + 63) / 64) * 64;

    float* w     = (float*)d_ws;
    float* diff  = w;
    float* disc  = w + QP;
    float* base  = w + 2 * QP;
    float* muT   = base + 0L * N_ROWS;
    float* lmdaT = base + 1L * N_ROWS;
    float* maskT = base + 2L * N_ROWS;
    float* epsT  = base + 3L * N_ROWS;
    float* A_ws  = base + 4L * N_ROWS;
    float* B_ws  = base + 5L * N_ROWS;
    float* abT   = base + 6L * N_ROWS;

    reparam_kernel<<<(Q + 255) / 256, 256, 0, stream>>>(diff_mu, diff_lv, eps_diff, diff, Q);
    reparam_kernel<<<(Q + 255) / 256, 256, 0, stream>>>(disc_mu, disc_lv, eps_disc, disc, Q);
    transpose_kernel<<<N_ROWS / 256, 256, 0, stream>>>(mask, eps_ab, maskT, epsT);
    mlp_wmma_kernel<<<4096, 256, 0, stream>>>(q_id, resp, diff, disc,
                                              W1, b1, W2, b2, W3, b3, muT, lmdaT);
    scan_kernel<<<U_DIM / 256, 256, 0, stream>>>(muT, lmdaT, maskT, epsT, A_ws, B_ws, abT);
    finalize_kernel<<<N_ROWS / 256, 256, 0, stream>>>(abT, q_id, diff, disc, (float*)d_out);
}
